// DiscRNNG_37357625540829
// MI455X (gfx1250) — compile-verified, matrix operand order checked by reading, not run on hardware
//
#include <hip/hip_runtime.h>

// =====================================================================
// DiscRNNG forward for MI455X (gfx1250, wave32, WMMA bf16 K=32)
//
// All GEMMs: M=64 (batch), N=128 or 512, K multiple of 32, computed as
// X @ W^T with W stored (N,K) row-major.  Every GEMM operand is stored
// bf16 at rest (activations converted once at production time in the
// producing kernel's epilogue), so WMMA fragments are pure b128 loads:
// no per-fragment conversion VALU on the critical path.  Accumulation
// is fp32; LSTM cell state c, biases, and all sigmoid/tanh math stay
// fp32; only tanh/sigmoid-bounded h and relu hiddens are bf16-rounded.
// =====================================================================

typedef __attribute__((ext_vector_type(16))) __bf16         bf16x16;
typedef __attribute__((ext_vector_type(16))) unsigned short ushort16;
typedef __attribute__((ext_vector_type(8)))  unsigned short ushort8;
typedef __attribute__((ext_vector_type(8)))  float          floatx8;

__device__ __forceinline__ unsigned short f2bf_rne(float f) {
  unsigned u = __float_as_uint(f);
  unsigned r = u + 0x7FFFu + ((u >> 16) & 1u);
  return (unsigned short)(r >> 16);
}
__device__ __forceinline__ float bf2f(unsigned short h) {
  return __uint_as_float(((unsigned)h) << 16);
}

// 16-bit fragment, ISA 16x32 layout: lane L holds row (A: m=L%16) or
// col (B: n=L%16); k = (v>>2)*16 + (L>>4)*8 + (v&3)*2 + j -> two
// contiguous 16-byte chunks at +0 and +16 elements from
// (row, kbase + (L>>4)*8).  Pure loads, no conversion.
__device__ __forceinline__ bf16x16 frag_bf16(const unsigned short* __restrict__ p) {
  const ushort8* q = (const ushort8*)p;
  ushort8 a0 = q[0], a1 = q[2];
  ushort16 u;
#pragma unroll
  for (int i = 0; i < 8; ++i) { u[i] = a0[i]; u[8 + i] = a1[i]; }
  return __builtin_bit_cast(bf16x16, u);
}

__device__ __forceinline__ floatx8 wmma_bf16(bf16x16 a, bf16x16 b, floatx8 c) {
  return __builtin_amdgcn_wmma_f32_16x16x32_bf16(false, a, false, b, (short)0, c,
                                                 false, false);
}

__device__ __forceinline__ float sigf(float x) { return 1.f / (1.f + expf(-x)); }

// ------------------------------------------------------------------
// Fused LSTM step: g = X@Wih^T + H@Whh^T + bih + bhh ; cell update.
// X (bf16, 64xK1, ld=ldx); Hin (bf16 64x128); Cin/Cout fp32; Hout bf16.
// Wih bf16 (512,K1); Whh bf16 (512,128).  first=1 => h,c == 0.
// 512 threads = 16 wave32; 32 (mtile, 16-col tile) combos; the 4 gate
// accumulators stay in registers through the sigmoid/tanh epilogue.
// ------------------------------------------------------------------
__global__ void lstm_step_kernel(const unsigned short* __restrict__ X, int ldx,
                                 const unsigned short* __restrict__ Wih, int K1,
                                 const unsigned short* __restrict__ Hin,
                                 const unsigned short* __restrict__ Whh,
                                 const float* __restrict__ Cin,
                                 const float* __restrict__ bih,
                                 const float* __restrict__ bhh,
                                 unsigned short* __restrict__ Hout,
                                 float* __restrict__ Cout, int first) {
  const int lane = threadIdx.x & 31;
  const int wave = threadIdx.x >> 5;  // 0..15
  const int lm = lane & 15;
  const int lh = lane >> 4;
  for (int cm = 0; cm < 2; ++cm) {
    const int combo = wave * 2 + cm;  // 0..31
    const int mt = combo >> 3;        // 0..3 (M tile)
    const int nt = combo & 7;         // 0..7 (16-col tile within 128)
    floatx8 acc[4];
#pragma unroll
    for (int q = 0; q < 4; ++q)
#pragma unroll
      for (int r = 0; r < 8; ++r) acc[q][r] = 0.f;

    {  // X @ Wih^T
      const unsigned short* xrow = X + (size_t)(mt * 16 + lm) * ldx + lh * 8;
      for (int kt = 0; kt < K1; kt += 32) {
        bf16x16 a = frag_bf16(xrow + kt);
#pragma unroll
        for (int q = 0; q < 4; ++q) {
          const unsigned short* wrow =
              Wih + (size_t)(q * 128 + nt * 16 + lm) * K1 + kt + lh * 8;
          acc[q] = wmma_bf16(a, frag_bf16(wrow), acc[q]);
        }
      }
    }
    if (!first) {  // H @ Whh^T
      const unsigned short* hrow = Hin + (size_t)(mt * 16 + lm) * 128 + lh * 8;
      for (int kt = 0; kt < 128; kt += 32) {
        bf16x16 a = frag_bf16(hrow + kt);
#pragma unroll
        for (int q = 0; q < 4; ++q) {
          const unsigned short* wrow =
              Whh + (size_t)(q * 128 + nt * 16 + lm) * 128 + kt + lh * 8;
          acc[q] = wmma_bf16(a, frag_bf16(wrow), acc[q]);
        }
      }
    }
    // epilogue: C/D layout lane n = lane&15, m = r + 8*(lane>>4)
    const int col = nt * 16 + lm;
#pragma unroll
    for (int r = 0; r < 8; ++r) {
      const int m = mt * 16 + r + lh * 8;
      float gi = acc[0][r] + bih[col] + bhh[col];
      float gf = acc[1][r] + bih[128 + col] + bhh[128 + col];
      float gg = acc[2][r] + bih[256 + col] + bhh[256 + col];
      float go = acc[3][r] + bih[384 + col] + bhh[384 + col];
      float c0 = first ? 0.f : Cin[m * 128 + col];
      float cn = sigf(gf) * c0 + sigf(gi) * tanhf(gg);
      float h  = sigf(go) * tanhf(cn);
      Hout[m * 128 + col] = f2bf_rne(h);
      Cout[m * 128 + col] = cn;
    }
  }
}

// ------------------------------------------------------------------
// Y(64,128 bf16) = act( X(64,K bf16) @ W^T + b ), W bf16 (128,K).
// ------------------------------------------------------------------
__global__ void linear128_kernel(const unsigned short* __restrict__ X, int ldx,
                                 const unsigned short* __restrict__ W, int K,
                                 const float* __restrict__ bias,
                                 unsigned short* __restrict__ Y, int relu) {
  const int lane = threadIdx.x & 31;
  const int wave = threadIdx.x >> 5;
  const int lm = lane & 15;
  const int lh = lane >> 4;
  for (int cm = 0; cm < 2; ++cm) {
    const int tile = wave * 2 + cm;  // 0..31
    const int mt = tile >> 3;
    const int nt = tile & 7;
    floatx8 acc;
#pragma unroll
    for (int r = 0; r < 8; ++r) acc[r] = 0.f;
    const unsigned short* xrow = X + (size_t)(mt * 16 + lm) * ldx + lh * 8;
    const unsigned short* wrow = W + (size_t)(nt * 16 + lm) * K + lh * 8;
    for (int kt = 0; kt < K; kt += 32) {
      if (kt + 32 < K) __builtin_prefetch(wrow + kt + 32, 0, 1);
      acc = wmma_bf16(frag_bf16(xrow + kt), frag_bf16(wrow + kt), acc);
    }
    const int col = nt * 16 + lm;
#pragma unroll
    for (int r = 0; r < 8; ++r) {
      const int m = mt * 16 + r + lh * 8;
      float v = acc[r] + bias[col];
      if (relu) v = fmaxf(v, 0.f);
      Y[m * 128 + col] = f2bf_rne(v);
    }
  }
}

// scores = hid @ W2^T + b2 (64x3), log-softmax, accumulate NLL mean.
__global__ void loss_step_kernel(const unsigned short* __restrict__ hid,
                                 const float* __restrict__ W2,
                                 const float* __restrict__ b2,
                                 const int* __restrict__ acts, int t, int TA,
                                 float* __restrict__ out) {
  __shared__ float red[64];
  const int b = threadIdx.x;  // 64 threads
  float s0 = b2[0], s1 = b2[1], s2 = b2[2];
  const unsigned short* hb = hid + b * 128;
  for (int k = 0; k < 128; ++k) {
    float x = bf2f(hb[k]);
    s0 += x * W2[k];
    s1 += x * W2[128 + k];
    s2 += x * W2[256 + k];
  }
  float m = fmaxf(s0, fmaxf(s1, s2));
  float lse = m + logf(expf(s0 - m) + expf(s1 - m) + expf(s2 - m));
  int tg = acts[b * TA + t];
  float sv = (tg == 0) ? s0 : ((tg == 1) ? s1 : s2);
  red[b] = sv - lse;
  __syncthreads();
  if (b == 0) {
    float sum = 0.f;
    for (int i = 0; i < 64; ++i) sum += red[i];
    out[0] -= sum * (1.f / 64.f);
  }
}

// out[(s*64+b)*D+d] = bf16( emb[idx[b*L + (rev? L-1-s : s)]*D + d] )
__global__ void gather_emb_kernel(const float* __restrict__ emb,
                                  const int* __restrict__ idx,
                                  unsigned short* __restrict__ out, int S,
                                  int L, int D, int rev) {
  long i = (long)blockIdx.x * blockDim.x + threadIdx.x;
  long total = (long)S * 64 * D;
  if (i >= total) return;
  int d = (int)(i % D);
  int b = (int)((i / D) & 63);
  int s = (int)(i / ((long)D * 64));
  int pos = rev ? (L - 1 - s) : s;
  int tok = idx[b * L + pos];
  out[i] = f2bf_rne(emb[(long)tok * D + d]);
}

__global__ void cvt_bf16_kernel(const float* __restrict__ src,
                                unsigned short* __restrict__ dst, int n) {
  int i = blockIdx.x * blockDim.x + threadIdx.x;
  if (i < n) dst[i] = f2bf_rne(src[i]);
}

// copy a (64,128) bf16 block (or broadcast a 128-vector) into columns
// [coloff, coloff+128) of a row-major (64, ldd) bf16 destination.
__global__ void copyseg_kernel(const unsigned short* __restrict__ src,
                               int bcast, unsigned short* __restrict__ dst,
                               int ldd, int coloff) {
  int i = blockIdx.x * blockDim.x + threadIdx.x;  // 8192
  if (i >= 64 * 128) return;
  int b = i >> 7, d = i & 127;
  unsigned short v = bcast ? src[d] : src[i];
  dst[b * ldd + coloff + d] = v;
}

__global__ void zero1_kernel(float* p) {
  if (threadIdx.x == 0 && blockIdx.x == 0) p[0] = 0.f;
}

// =====================================================================
// Host orchestration.  Inputs flattened depth-first in insertion order:
//  0 words(64x64) 1 nonterms(64x17) 2 actions(64x98)
//  3 word_emb 4 nt_emb 5 act_emb
//  6/7 buffer2stack W,b    8/9 nt2stack W,b
// 10..17 subtree_fwd L0{Wih,Whh,bih,bhh} L1{...}
// 18..25 subtree_bwd      26/27 subtree_proj1  28/29 subtree_proj2
// 30..37 buffer_lstm      38..45 history_lstm  46..53 stack_lstm
// 54/55 action_proj1      56/57 action_proj2
// 58 buffer_guard 59 history_guard 60 stack_guard
// =====================================================================

struct LSTMW {
  const unsigned short* Wih;
  int K;
  const unsigned short* Whh;
  const float* bih;
  const float* bhh;
};

extern "C" void kernel_launch(void* const* d_in, const int* in_sizes, int n_in,
                              void* d_out, int out_size, void* d_ws,
                              size_t ws_size, hipStream_t stream) {
  (void)in_sizes; (void)n_in; (void)out_size; (void)ws_size;

  const int* words    = (const int*)d_in[0];
  const int* nonterms = (const int*)d_in[1];
  const int* actions  = (const int*)d_in[2];
  const float* word_emb = (const float*)d_in[3];
  const float* nt_emb   = (const float*)d_in[4];
  const float* act_emb  = (const float*)d_in[5];

  // ---- workspace bump allocator (deterministic) ----
  char* wp = (char*)d_ws;
  auto alloc_f = [&](size_t n) -> float* {
    float* p = (float*)wp;
    wp += ((n * 4 + 255) / 256) * 256;
    return p;
  };
  auto alloc_h = [&](size_t n) -> unsigned short* {
    unsigned short* p = (unsigned short*)wp;
    wp += ((n * 2 + 255) / 256) * 256;
    return p;
  };

  const int SB = 64 * 128;  // one (64,128) state block
  unsigned short* buff   = alloc_h((size_t)64 * 64 * 512);  // reversed word emb
  unsigned short* ntb    = alloc_h((size_t)17 * SB);        // reversed NT emb
  unsigned short* histin = alloc_h((size_t)98 * SB);
  unsigned short* benc   = alloc_h((size_t)64 * SB);
  unsigned short* henc   = alloc_h((size_t)98 * SB);
  unsigned short* slots  = alloc_h((size_t)24 * SB);        // stack (max 21)
  unsigned short* h0[2] = {alloc_h(SB), alloc_h(SB)};
  unsigned short* h1[2] = {alloc_h(SB), alloc_h(SB)};
  float* c0[2] = {alloc_f(SB), alloc_f(SB)};  // cell state stays fp32
  float* c1[2] = {alloc_f(SB), alloc_f(SB)};
  unsigned short* xcat = alloc_h((size_t)64 * 384);
  unsigned short* hid1 = alloc_h(SB);

  // ---- weight/guard conversion fp32 -> bf16 (per call; no cached state)
  auto cvt = [&](const float* s, size_t n) -> const unsigned short* {
    unsigned short* d = alloc_h(n);
    cvt_bf16_kernel<<<(int)((n + 255) / 256), 256, 0, stream>>>(s, d, (int)n);
    return d;
  };
  LSTMW FWD[2], BWD[2], BUF[2], HIS[2], STK[2];
  auto mklstm = [&](int base, int K0, LSTMW* o) {
    for (int l = 0; l < 2; ++l) {
      int K = (l == 0) ? K0 : 128;
      o[l].Wih = cvt((const float*)d_in[base + l * 4 + 0], (size_t)512 * K);
      o[l].Whh = cvt((const float*)d_in[base + l * 4 + 1], (size_t)512 * 128);
      o[l].bih = (const float*)d_in[base + l * 4 + 2];
      o[l].bhh = (const float*)d_in[base + l * 4 + 3];
      o[l].K = K;
    }
  };
  mklstm(10, 128, FWD);
  mklstm(18, 128, BWD);
  mklstm(30, 512, BUF);
  mklstm(38, 128, HIS);
  mklstm(46, 128, STK);
  const unsigned short* b2sW = cvt((const float*)d_in[6], (size_t)128 * 512);
  const float* b2sb = (const float*)d_in[7];
  const unsigned short* n2sW = cvt((const float*)d_in[8], (size_t)128 * 128);
  const float* n2sb = (const float*)d_in[9];
  const unsigned short* p1W = cvt((const float*)d_in[26], (size_t)128 * 256);
  const float* p1b = (const float*)d_in[27];
  const unsigned short* p2W = cvt((const float*)d_in[28], (size_t)128 * 128);
  const float* p2b = (const float*)d_in[29];
  const unsigned short* a1W = cvt((const float*)d_in[54], (size_t)128 * 384);
  const float* a1b = (const float*)d_in[55];
  const float* a2W = (const float*)d_in[56];  // (3,128) stays fp32
  const float* a2b = (const float*)d_in[57];
  const unsigned short* bguard = cvt((const float*)d_in[58], 128);
  const unsigned short* hguard = cvt((const float*)d_in[59], 128);
  const unsigned short* sguard = cvt((const float*)d_in[60], 128);

  // ---- launch helpers ----
  auto lstm = [&](const unsigned short* X, int ldx, const LSTMW& w,
                  const unsigned short* Hin, const float* Cin,
                  unsigned short* Hout, float* Cout, int first) {
    lstm_step_kernel<<<1, 512, 0, stream>>>(X, ldx, w.Wih, w.K, Hin, w.Whh,
                                            Cin, w.bih, w.bhh, Hout, Cout,
                                            first);
  };
  auto lin = [&](const unsigned short* X, int ldx, const unsigned short* W,
                 int K, const float* b, unsigned short* Y, int relu) {
    linear128_kernel<<<1, 512, 0, stream>>>(X, ldx, W, K, b, Y, relu);
  };
  auto cpseg = [&](const unsigned short* src, int bc, unsigned short* dst,
                   int ldd, int off) {
    copyseg_kernel<<<32, 256, 0, stream>>>(src, bc, dst, ldd, off);
  };

  // ---- loss accumulator init (graph-replay safe) ----
  zero1_kernel<<<1, 32, 0, stream>>>((float*)d_out);

  // ---- embedding gathers (convert to bf16 once here) ----
  {
    int tot = 64 * 64 * 512;
    gather_emb_kernel<<<(tot + 255) / 256, 256, 0, stream>>>(
        word_emb, words, buff, 64, 64, 512, 1);
  }
  {
    int tot = 17 * 64 * 128;
    gather_emb_kernel<<<(tot + 255) / 256, 256, 0, stream>>>(
        nt_emb, nonterms, ntb, 17, 17, 128, 1);
  }
  {
    int tot = 98 * 64 * 128;
    gather_emb_kernel<<<(tot + 255) / 256, 256, 0, stream>>>(
        act_emb, actions, histin, 98, 98, 128, 0);
  }

  // ---- buffer encoder (2-layer LSTM over 64 reversed steps) ----
  {
    int p = 0;
    for (int s = 0; s < 64; ++s) {
      lstm(buff + (size_t)s * 64 * 512, 512, BUF[0], h0[p], c0[p], h0[p ^ 1],
           c0[p ^ 1], s == 0);
      const unsigned short* h1in = s ? (benc + (size_t)(s - 1) * SB) : h1[p];
      lstm(h0[p ^ 1], 128, BUF[1], h1in, c1[p], benc + (size_t)s * SB,
           c1[p ^ 1], s == 0);
      p ^= 1;
    }
  }
  // ---- history encoder (98 steps) ----
  {
    int p = 0;
    for (int s = 0; s < 98; ++s) {
      lstm(histin + (size_t)s * SB, 128, HIS[0], h0[p], c0[p], h0[p ^ 1],
           c0[p ^ 1], s == 0);
      const unsigned short* h1in = s ? (henc + (size_t)(s - 1) * SB) : h1[p];
      lstm(h0[p ^ 1], 128, HIS[1], h1in, c1[p], henc + (size_t)s * SB,
           c1[p ^ 1], s == 0);
      p ^= 1;
    }
  }

  // run a 2-layer LSTM over a slot-index sequence; returns last h of L1
  auto run_seq = [&](const LSTMW* L, const int* seq,
                     int n) -> const unsigned short* {
    int p = 0;
    for (int s = 0; s < n; ++s) {
      lstm(slots + (size_t)seq[s] * SB, 128, L[0], h0[p], c0[p], h0[p ^ 1],
           c0[p ^ 1], s == 0);
      lstm(h0[p ^ 1], 128, L[1], h1[p], c1[p], h1[p ^ 1], c1[p ^ 1], s == 0);
      p ^= 1;
    }
    return h1[p];
  };

  // ---- static action schedule (SLEN=64, CHUNK=4): [2] + 16*[2,1,1,1,1,0] + [0]
  int acts[98];
  {
    int na = 0;
    acts[na++] = 2;
    for (int c = 0; c < 16; ++c) {
      acts[na++] = 2;
      for (int i = 0; i < 4; ++i) acts[na++] = 1;
      acts[na++] = 0;
    }
    acts[na++] = 0;
  }

  int buff_len = 64, nt_len = 17;
  int stack_n = 0;
  bool open_[32];
  int idxseq[24];

  for (int t = 0; t < 98; ++t) {
    // --- action scores + loss ---
    const unsigned short* be =
        (buff_len > 0) ? (benc + (size_t)(buff_len - 1) * SB) : bguard;
    int beb = (buff_len > 0) ? 0 : 1;
    const unsigned short* he = (t > 0) ? (henc + (size_t)(t - 1) * SB) : hguard;
    int heb = (t > 0) ? 0 : 1;
    const unsigned short* se;
    int seb;
    if (stack_n > 0) {
      for (int i = 0; i < stack_n; ++i) idxseq[i] = i;
      se = run_seq(STK, idxseq, stack_n);
      seb = 0;
    } else {
      se = sguard;
      seb = 1;
    }
    cpseg(be, beb, xcat, 384, 0);
    cpseg(he, heb, xcat, 384, 128);
    cpseg(se, seb, xcat, 384, 256);
    lin(xcat, 384, a1W, 384, a1b, hid1, 1);
    loss_step_kernel<<<1, 64, 0, stream>>>(hid1, a2W, a2b, actions, t, 98,
                                           (float*)d_out);

    // --- apply action ---
    int a = acts[t];
    if (a == 0) {  // REDUCE
      int ch[24], nch = 0;
      while (stack_n > 0 && !open_[stack_n - 1]) ch[nch++] = --stack_n;
      int parent = --stack_n;
      // forward: [parent, top, top-1, ...]
      idxseq[0] = parent;
      for (int i = 0; i < nch; ++i) idxseq[1 + i] = ch[i];
      const unsigned short* of = run_seq(FWD, idxseq, nch + 1);
      cpseg(of, 0, xcat, 256, 0);
      // backward: [parent, bottom-child, ..., top]
      idxseq[0] = parent;
      for (int i = 0; i < nch; ++i) idxseq[1 + i] = ch[nch - 1 - i];
      const unsigned short* ob = run_seq(BWD, idxseq, nch + 1);
      cpseg(ob, 0, xcat, 256, 128);
      lin(xcat, 256, p1W, 256, p1b, hid1, 1);
      lin(hid1, 128, p2W, 128, p2b, slots + (size_t)parent * SB, 0);
      open_[parent] = false;
      stack_n = parent + 1;
    } else if (a == 1) {  // SHIFT
      lin(buff + (size_t)(buff_len - 1) * 64 * 512, 512, b2sW, 512, b2sb,
          slots + (size_t)stack_n * SB, 0);
      open_[stack_n] = false;
      ++stack_n;
      --buff_len;
    } else {  // NT
      lin(ntb + (size_t)(nt_len - 1) * SB, 128, n2sW, 128, n2sb,
          slots + (size_t)stack_n * SB, 0);
      open_[stack_n] = true;
      ++stack_n;
      --nt_len;
    }
  }
}
